// MiddleModel_64441689309574
// MI455X (gfx1250) — matrix operand order repaired and checked by reading, not compiled, not run on hardware
//
#include <hip/hip_runtime.h>
#include <hip/hip_bf16.h>

// ---------------------------------------------------------------------------
// PointNet-seg forward for MI455X (gfx1250, wave32).
// GEMMs: bf16-staged operands -> v_wmma_f32_16x16x32_bf16, 32x32 C tile/wave,
// b128 fragment loads, f32 accumulate. Elementwise/graph ops: VALU + f32
// global atomics.
// ---------------------------------------------------------------------------

typedef __attribute__((ext_vector_type(16))) __bf16 v16bf;
typedef __attribute__((ext_vector_type(8)))  __bf16 v8bf;
typedef __attribute__((ext_vector_type(8)))  float  v8f;

static constexpr int   Bsz = 16, Npt = 2048, BNt = Bsz * Npt;
static constexpr int   KNN = 20, NCls = 50;
static constexpr long  E0  = (long)BNt * KNN;      // 655360 edges per kNN list
static constexpr float EPSF = 1e-5f;

#define DEV __device__ __forceinline__

DEV __bf16 f2bf(float f) {                          // round-to-nearest-even
  unsigned u = __float_as_uint(f);
  unsigned r = u + 0x7FFFu + ((u >> 16) & 1u);
  unsigned short h = (unsigned short)(r >> 16);
  return __builtin_bit_cast(__bf16, h);
}

DEV v8bf ld8(const __bf16* p) { return *(const v8bf*)p; }   // 16B aligned -> b128

DEV v16bf cat16(v8bf lo, v8bf hi) {
  return __builtin_shufflevector(lo, hi, 0, 1, 2, 3, 4, 5, 6, 7,
                                 8, 9, 10, 11, 12, 13, 14, 15);
}

// ---------------------------------------------------------------------------
// WMMA GEMM, bf16 operands: C[b] = A[b] * Bt[b]^T (+bias).
//   A : [M,K] bf16 row-major, Bt : [N,K] bf16 row-major (contiguous K loads).
// Requires K%32==0, N%32==0.  M edge handled by clamped loads+guarded stores.
// 256 threads = 8 waves (2Mx4N); block tile 64x128; wave tile 32x32 (4 accs).
// Per 32-K step: 8 x global_load_b128, 4 x v_wmma  (A/B fragments reused 2x).
// ---------------------------------------------------------------------------
__global__ void k_gemm_wmma_bf(const __bf16* __restrict__ A, const __bf16* __restrict__ Bt,
                               float* __restrict__ C, __bf16* __restrict__ Cb,
                               const float* __restrict__ bias,
                               int M, int N, int K, int lda, int ldb, int ldc,
                               long sA, long sB, long sC)
{
  const int lane = threadIdx.x & 31;
  const int wave = threadIdx.x >> 5;
  const int m0 = blockIdx.y * 64 + (wave & 1) * 32;
  const int n0 = blockIdx.x * 128 + (wave >> 1) * 32;
  if (m0 >= M || n0 >= N) return;                  // wave-uniform, EXEC stays all-1s
  A  += (long)blockIdx.z * sA;
  Bt += (long)blockIdx.z * sB;
  C  += (long)blockIdx.z * sC;
  if (Cb) Cb += (long)blockIdx.z * sC;

  const int l15 = lane & 15, kg = lane >> 4;
  // A fragment rows (clamped so partial-M tiles read in-bounds; results for
  // clamped rows are discarded at the store).
  int am0 = m0 + l15;      if (am0 > M - 1) am0 = M - 1;
  int am1 = m0 + 16 + l15; if (am1 > M - 1) am1 = M - 1;
  const int ab = kg * 8;      // A half-layout: elems 0..7 -> K+ab, 8..15 -> K+16+ab
  const int bb = kg * 8;      // B half-layout: elems 0..7 -> K+bb, 8..15 -> K+16+bb
  const __bf16* A0 = A + (long)am0 * lda + ab;
  const __bf16* A1 = A + (long)am1 * lda + ab;
  const __bf16* B0 = Bt + (long)(n0 + l15) * ldb + bb;
  const __bf16* B1 = Bt + (long)(n0 + 16 + l15) * ldb + bb;

  v8f c00 = {}, c01 = {}, c10 = {}, c11 = {};
  for (int kk = 0; kk < K; kk += 32) {
    __builtin_prefetch(A0 + kk + 512, 0, 0);       // global_prefetch_b8
    __builtin_prefetch(B0 + kk + 512, 0, 0);
    v16bf a0 = cat16(ld8(A0 + kk), ld8(A0 + kk + 16));
    v16bf a1 = cat16(ld8(A1 + kk), ld8(A1 + kk + 16));
    v16bf b0 = cat16(ld8(B0 + kk), ld8(B0 + kk + 16));
    v16bf b1 = cat16(ld8(B1 + kk), ld8(B1 + kk + 16));
    c00 = __builtin_amdgcn_wmma_f32_16x16x32_bf16(false, a0, false, b0, (short)0, c00, false, false);
    c01 = __builtin_amdgcn_wmma_f32_16x16x32_bf16(false, a0, false, b1, (short)0, c01, false, false);
    c10 = __builtin_amdgcn_wmma_f32_16x16x32_bf16(false, a1, false, b0, (short)0, c10, false, false);
    c11 = __builtin_amdgcn_wmma_f32_16x16x32_bf16(false, a1, false, b1, (short)0, c11, false, false);
  }
  // C/D layout: VGPR v, lane l -> row v + 8*(l>>4), col (l&15).
  const int cn0 = n0 + l15, cn1 = n0 + 16 + l15;
  const float bv0 = bias ? bias[cn0] : 0.0f;
  const float bv1 = bias ? bias[cn1] : 0.0f;
#pragma unroll
  for (int v = 0; v < 8; ++v) {
    const int r0 = m0 + v + 8 * kg;
    const int r1 = r0 + 16;
    if (r0 < M) {
      float x0 = c00[v] + bv0, x1 = c01[v] + bv1;
      C[(long)r0 * ldc + cn0] = x0;
      C[(long)r0 * ldc + cn1] = x1;
      if (Cb) { Cb[(long)r0 * ldc + cn0] = f2bf(x0); Cb[(long)r0 * ldc + cn1] = f2bf(x1); }
    }
    if (r1 < M) {
      float x0 = c10[v] + bv0, x1 = c11[v] + bv1;
      C[(long)r1 * ldc + cn0] = x0;
      C[(long)r1 * ldc + cn1] = x1;
      if (Cb) { Cb[(long)r1 * ldc + cn0] = f2bf(x0); Cb[(long)r1 * ldc + cn1] = f2bf(x1); }
    }
  }
}

// Small / ragged GEMM (K=3 or N=50 paths) - one thread per output element.
template <bool BT>
__global__ void k_gemm_small(const float* __restrict__ A, const float* __restrict__ B,
                             float* __restrict__ C, const float* __restrict__ bias,
                             int M, int N, int K, int lda, int ldb, int ldc)
{
  long t = (long)blockIdx.x * blockDim.x + threadIdx.x;
  if (t >= (long)M * N) return;
  int m = (int)(t / N), n = (int)(t % N);
  float s = bias ? bias[n] : 0.0f;
  for (int k = 0; k < K; ++k)
    s += A[(long)m * lda + k] * (BT ? B[(long)n * ldb + k] : B[(long)k * ldb + n]);
  C[(long)m * ldc + n] = s;
}

// fp32 -> bf16 (weights)
__global__ void k_cvt(const float* __restrict__ X, __bf16* __restrict__ Y, long n)
{
  long t = (long)blockIdx.x * blockDim.x + threadIdx.x;
  if (t < n) Y[t] = f2bf(X[t]);
}

// tk[b, k*128+n] -> tkT_bf[b, n*128+k]   (to the BT layout used by the GEMM)
__global__ void k_cvtT(const float* __restrict__ X, __bf16* __restrict__ Y)
{
  long t = (long)blockIdx.x * blockDim.x + threadIdx.x;
  if (t >= (long)Bsz * 128 * 128) return;
  int b = (int)(t >> 14), r = (int)(t & 16383);
  int k = r >> 7, n = r & 127;
  Y[((long)b << 14) + n * 128 + k] = f2bf(X[t]);
}

// Per-channel mean/var of X[R,C] (training-mode BN statistics).
__global__ void k_colstats(const float* __restrict__ X, float* mu, float* var, int R, int C)
{
  int c = blockIdx.x;
  float s = 0.f, s2 = 0.f;
  for (int r = threadIdx.x; r < R; r += blockDim.x) {
    float v = X[(long)r * C + c];
    s += v; s2 += v * v;
  }
  __shared__ float sh[256], sh2[256];
  sh[threadIdx.x] = s; sh2[threadIdx.x] = s2;
  __syncthreads();
  for (int o = blockDim.x >> 1; o > 0; o >>= 1) {
    if (threadIdx.x < (unsigned)o) { sh[threadIdx.x] += sh[threadIdx.x + o]; sh2[threadIdx.x] += sh2[threadIdx.x + o]; }
    __syncthreads();
  }
  if (threadIdx.x == 0) {
    float m = sh[0] / R;
    mu[c] = m;
    var[c] = sh2[0] / R - m * m;
  }
}

// BN (+optional ReLU), fp32 in place + optional fused bf16 copy for GEMM use.
__global__ void k_bn_act(float* __restrict__ X, __bf16* __restrict__ Xb,
                         const float* g, const float* b,
                         const float* mu, const float* var, int C, long n, int relu)
{
  long t = (long)blockIdx.x * blockDim.x + threadIdx.x;
  if (t >= n) return;
  int c = (int)(t % C);
  float v = (X[t] - mu[c]) * rsqrtf(var[c] + EPSF) * g[c] + b[c];
  if (relu) v = fmaxf(v, 0.f);
  X[t] = v;
  if (Xb) Xb[t] = f2bf(v);
}

__global__ void k_maxpool(const float* __restrict__ X, float* __restrict__ O,
                          __bf16* __restrict__ Ob, int C)
{ // X:[B*N,C] -> O:[B,C]   (global max pool over points)
  int t = blockIdx.x * blockDim.x + threadIdx.x;
  if (t >= Bsz * C) return;
  int b = t / C, c = t % C;
  const float* p = X + (long)b * Npt * C + c;
  float m = -3.4e38f;
  for (int n = 0; n < Npt; ++n) m = fmaxf(m, p[(long)n * C]);
  O[t] = m;
  Ob[t] = f2bf(m);
}

__global__ void k_add_eye(float* M, int c, int cc)
{
  int t = blockIdx.x * blockDim.x + threadIdx.x;
  if (t >= Bsz * c) return;
  int b = t / c, i = t % c;
  M[(long)b * cc + i * c + i] += 1.0f;
}

__global__ void k_transform3(const float* __restrict__ X, const float* __restrict__ T,
                             float* __restrict__ H)
{ // H[bn,d] = sum_c X[bn,c] * T[b,c,d]
  int t = blockIdx.x * blockDim.x + threadIdx.x;
  if (t >= BNt) return;
  int b = t / Npt;
  const float* x = X + (long)t * 3;
  const float* m = T + b * 9;
#pragma unroll
  for (int d = 0; d < 3; ++d)
    H[(long)t * 3 + d] = x[0] * m[d] + x[1] * m[3 + d] + x[2] * m[6 + d];
}

// kNN on raw xyz (K=3): direct distance scan, exact top-k with index tiebreak.
__global__ void k_knn3(const float* __restrict__ X, int* __restrict__ row, int* __restrict__ col)
{
  int i = blockIdx.x * blockDim.x + threadIdx.x;
  if (i >= BNt) return;
  int b = i / Npt, il = i - b * Npt;
  const float* xb = X + (long)b * Npt * 3;
  float x0 = xb[il * 3], x1 = xb[il * 3 + 1], x2 = xb[il * 3 + 2];
  float bd[KNN]; int bi[KNN];
  for (int t = 0; t < KNN; ++t) { bd[t] = 3.4e38f; bi[t] = 0; }
  for (int j = 0; j < Npt; ++j) {
    if (j == il) continue;
    float d0 = xb[j * 3] - x0, d1 = xb[j * 3 + 1] - x1, d2 = xb[j * 3 + 2] - x2;
    float d = d0 * d0 + d1 * d1 + d2 * d2;
    if (d < bd[KNN - 1]) {
      int p = KNN - 1;
      while (p > 0 && bd[p - 1] > d) { bd[p] = bd[p - 1]; bi[p] = bi[p - 1]; --p; }
      bd[p] = d; bi[p] = j;
    }
  }
  long e = (long)i * KNN;
  for (int t = 0; t < KNN; ++t) { row[e + t] = b * Npt + bi[t]; col[e + t] = i; }
}

__global__ void k_sqnorm(const float* __restrict__ X, float* __restrict__ sq, int C)
{
  int i = blockIdx.x * blockDim.x + threadIdx.x;
  if (i >= Npt) return;
  const float* p = X + (long)i * C;
  float s = 0.f;
  for (int c = 0; c < C; ++c) s += p[c] * p[c];
  sq[i] = s;
}

// top-k over one batch's precomputed Gram matrix: d2 = sq_i + sq_j - 2*gram.
__global__ void k_topk_gram(const float* __restrict__ gram, const float* __restrict__ sq,
                            int b, int* __restrict__ row, int* __restrict__ col)
{
  int i = blockIdx.x * blockDim.x + threadIdx.x;
  if (i >= Npt) return;
  float si = sq[i];
  float bd[KNN]; int bi[KNN];
  for (int t = 0; t < KNN; ++t) { bd[t] = 3.4e38f; bi[t] = 0; }
  const float* gr = gram + (long)i * Npt;
  for (int j = 0; j < Npt; ++j) {
    if (j == i) continue;
    float d = si + sq[j] - 2.0f * gr[j];
    if (d < bd[KNN - 1]) {
      int p = KNN - 1;
      while (p > 0 && bd[p - 1] > d) { bd[p] = bd[p - 1]; bi[p] = bi[p - 1]; --p; }
      bd[p] = d; bi[p] = j;
    }
  }
  long e = E0 + ((long)(b * Npt + i)) * KNN;
  for (int t = 0; t < KNN; ++t) { row[e + t] = b * Npt + bi[t]; col[e + t] = b * Npt + i; }
}

// first_mask: within each kNN list pairs are unique, so duplicates only arise
// when a list-2 edge already exists in list-1 (stable sort keeps list-1 first).
__global__ void k_mask2(const int* __restrict__ row, const int* __restrict__ col,
                        int* __restrict__ mask)
{
  long e = (long)blockIdx.x * blockDim.x + threadIdx.x;
  if (e >= E0) return;
  int r = row[E0 + e], c = col[E0 + e];
  long base = (long)c * KNN;                 // list-1 edges of the same center
  int dup = 0;
  for (int t = 0; t < KNN; ++t) dup |= (row[base + t] == r);
  mask[E0 + e] = dup ? 0 : 1;
}

__global__ void k_edgecos(const int* __restrict__ row, const int* __restrict__ col,
                          const float* __restrict__ nrm, float* __restrict__ cv, long E)
{
  long e = (long)blockIdx.x * blockDim.x + threadIdx.x;
  if (e >= E) return;
  const float* a = nrm + (long)row[e] * 3;
  const float* b = nrm + (long)col[e] * 3;
  float na = sqrtf(a[0] * a[0] + a[1] * a[1] + a[2] * a[2]);
  float nb = sqrtf(b[0] * b[0] + b[1] * b[1] + b[2] * b[2]);
  float den = fmaxf(na * nb, 1e-8f);
  cv[e] = fabsf((a[0] * b[0] + a[1] * b[1] + a[2] * b[2]) / den);
}

__global__ void k_argmax_row(const float* __restrict__ X, int* __restrict__ out, int C, int R)
{
  int r = blockIdx.x * blockDim.x + threadIdx.x;
  if (r >= R) return;
  const float* p = X + (long)r * C;
  float m = p[0]; int mi = 0;
  for (int c = 1; c < C; ++c) if (p[c] > m) { m = p[c]; mi = c; }
  out[r] = mi;
}

__global__ void k_vote(const int* __restrict__ row, const int* __restrict__ col,
                       const int* __restrict__ mask, const int* __restrict__ label,
                       float* __restrict__ res, long E)
{
  long e = (long)blockIdx.x * blockDim.x + threadIdx.x;
  if (e >= E) return;
  if (mask[e]) atomicAdd(&res[(long)col[e] * NCls + label[row[e]]], 1.0f);
}

__global__ void k_valid_w(const int* __restrict__ row, const int* __restrict__ col,
                          const int* __restrict__ mask, const int* __restrict__ label,
                          const int* __restrict__ freq, const float* __restrict__ cv,
                          float* __restrict__ w, long E)
{
  long e = (long)blockIdx.x * blockDim.x + threadIdx.x;
  if (e >= E) return;
  w[e] = (mask[e] && label[row[e]] == freq[col[e]]) ? cv[e] : 0.0f;
}

__global__ void k_deg(const int* __restrict__ col, const float* __restrict__ w,
                      float* __restrict__ deg, long E)
{
  long e = (long)blockIdx.x * blockDim.x + threadIdx.x;
  if (e >= E) return;
  atomicAdd(&deg[col[e]], w[e]);
}

__global__ void k_dis(const float* __restrict__ deg, float* __restrict__ dis, int n)
{
  int i = blockIdx.x * blockDim.x + threadIdx.x;
  if (i >= n) return;
  float d = deg[i];
  dis[i] = d > 0.f ? rsqrtf(fmaxf(d, 1e-12f)) : 0.f;
}

__global__ void k_gcn_init(const float* __restrict__ xw, const float* __restrict__ dis,
                           const float* __restrict__ bias, float* __restrict__ gout)
{ // self-loop (ww=1): out[n,:] = b + dis[n]^2 * xw[n,:]
  long t = (long)blockIdx.x * blockDim.x + threadIdx.x;
  if (t >= (long)BNt * NCls) return;
  int n = (int)(t / NCls), j = (int)(t % NCls);
  gout[t] = bias[j] + dis[n] * dis[n] * xw[t];
}

__global__ void k_gcn_edge(const int* __restrict__ row, const int* __restrict__ col,
                           const float* __restrict__ w, const float* __restrict__ dis,
                           const float* __restrict__ xw, float* __restrict__ gout, long E)
{
  long e = (long)blockIdx.x * blockDim.x + threadIdx.x;
  if (e >= E) return;
  float wv = w[e];
  if (wv == 0.f) return;
  int r = row[e], c = col[e];
  float nw = dis[r] * wv * dis[c];
  const float* xr = xw + (long)r * NCls;
  float* oc = gout + (long)c * NCls;
  for (int j = 0; j < NCls; ++j) atomicAdd(oc + j, nw * xr[j]);
}

__global__ void k_softmax_out(const float* __restrict__ G, float* __restrict__ lg,
                              float* __restrict__ pr, float* __restrict__ lp)
{
  int r = blockIdx.x * blockDim.x + threadIdx.x;
  if (r >= BNt) return;
  const float* g = G + (long)r * NCls;
  float m = -3.4e38f;
  for (int j = 0; j < NCls; ++j) m = fmaxf(m, g[j]);
  float s = 0.f;
  for (int j = 0; j < NCls; ++j) s += expf(g[j] - m);
  float ls = logf(s);
  for (int j = 0; j < NCls; ++j) {
    float v = g[j];
    lg[(long)r * NCls + j] = v;
    pr[(long)r * NCls + j] = expf(v - m) / s;
    lp[(long)r * NCls + j] = v - m - ls;
  }
}

__global__ void k_write_bcn(const float* __restrict__ src, float* __restrict__ dst,
                            int Cs, int Ctot, int co)
{ // src[(b*N+n)*Cs + c] -> dst[b*Ctot*N + (co+c)*N + n]
  long t = (long)blockIdx.x * blockDim.x + threadIdx.x;
  if (t >= (long)BNt * Cs) return;
  int bn = (int)(t / Cs), c = (int)(t % Cs);
  int b = bn / Npt, n = bn % Npt;
  dst[((long)b * Ctot + co + c) * Npt + n] = src[t];
}

__global__ void k_fill_f(float* p, float v, long n)
{ long t = (long)blockIdx.x * blockDim.x + threadIdx.x; if (t < n) p[t] = v; }
__global__ void k_fill_i(int* p, int v, long n)
{ long t = (long)blockIdx.x * blockDim.x + threadIdx.x; if (t < n) p[t] = v; }

// ---------------------------------------------------------------------------
// Host orchestration
// ---------------------------------------------------------------------------
#define G1(n) dim3((unsigned)(((n) + 255) / 256)), dim3(256)

static void gemmB(hipStream_t s, const __bf16* A, const __bf16* Bt, float* C, __bf16* Cb,
                  const float* bias, int M, int N, int K, int lda, int ldb, int ldc,
                  int batch = 1, long sA = 0, long sB = 0, long sC = 0)
{
  dim3 g((N + 127) / 128, (M + 63) / 64, batch);
  k_gemm_wmma_bf<<<g, 256, 0, s>>>(A, Bt, C, Cb, bias, M, N, K, lda, ldb, ldc, sA, sB, sC);
}

// Input pytree flattening order (dict keys sorted, tuples in order):
enum {
  I_X = 0, I_NORM = 1,
  I_B1 = 2, /*b2..b5*/
  I_BN1G = 7,  I_BN1B, I_BN2G, I_BN2B, I_BN3G, I_BN3B, I_BN4G, I_BN4B, I_BN5G, I_BN5B,
  I_CLS_B1 = 17, I_CLS_B2, I_CLS_B3, I_CLS_BN1G, I_CLS_BN1B, I_CLS_BN2G, I_CLS_BN2B,
  I_CLS_W1, I_CLS_W2, I_CLS_W3,
  I_GCN_W = 27, I_GCN_B = 28,
  I_T3 = 29,     // tnet3 block base (22 leaves)
  I_TK = 51,     // tnetk block base (22 leaves)
  I_W1 = 73, I_W2, I_W3, I_W4, I_W5
};
// offsets inside a tnet block:
enum { T_B1 = 0, T_B2, T_B3, T_BN1G, T_BN1B, T_BN2G, T_BN2B, T_BN3G, T_BN3B,
       T_BN4G, T_BN4B, T_BN5G, T_BN5B, T_FB1, T_FB2, T_FB3, T_FW1, T_FW2, T_FW3,
       T_W1, T_W2, T_W3 };

extern "C" void kernel_launch(void* const* d_in, const int* in_sizes, int n_in,
                              void* d_out, int out_size, void* d_ws, size_t ws_size,
                              hipStream_t stream)
{
  (void)in_sizes; (void)n_in; (void)out_size; (void)ws_size;
  auto F = [&](int i) { return (const float*)d_in[i]; };
  hipStream_t s = stream;

  // ---- workspace carve-out -------------------------------------------------
  char* ws = (char*)d_ws; size_t off = 0;
  auto alloc = [&](size_t bytes) { void* p = ws + off; off = (off + bytes + 255) & ~(size_t)255; return p; };
  // fp32 activations (BN stats / outputs read these)
  float* h64   = (float*)alloc((size_t)BNt * 64 * 4);
  float* h128  = (float*)alloc((size_t)BNt * 128 * 4);
  float* h1024 = (float*)alloc((size_t)BNt * 1024 * 4);
  float* pool  = (float*)alloc((size_t)Bsz * 1024 * 4);
  float* fcA   = (float*)alloc((size_t)Bsz * 512 * 4);
  float* fcB   = (float*)alloc((size_t)Bsz * 256 * 4);
  float* t3m   = (float*)alloc((size_t)Bsz * 9 * 4);
  float* tkm   = (float*)alloc((size_t)Bsz * 16384 * 4);
  float* h3    = (float*)alloc((size_t)BNt * 3 * 4);
  float* out1  = (float*)alloc((size_t)BNt * 64 * 4);
  float* out2  = (float*)alloc((size_t)BNt * 128 * 4);
  float* out3  = (float*)alloc((size_t)BNt * 128 * 4);
  float* out4  = (float*)alloc((size_t)BNt * 256 * 4);
  float* out5  = (float*)alloc((size_t)BNt * 512 * 4);
  float* mu    = (float*)alloc(2048 * 4);
  float* var   = (float*)alloc(2048 * 4);
  float* sq    = (float*)alloc((size_t)Npt * 4);
  float* gram  = (float*)alloc((size_t)Npt * Npt * 4);
  int*   erow  = (int*)alloc((size_t)2 * E0 * 4);
  int*   ecol  = (int*)alloc((size_t)2 * E0 * 4);
  int*   emask = (int*)alloc((size_t)2 * E0 * 4);
  float* ecos  = (float*)alloc((size_t)2 * E0 * 4);
  float* ew    = (float*)alloc((size_t)2 * E0 * 4);
  float* c256  = (float*)alloc((size_t)BNt * 256 * 4);
  float* c128  = (float*)alloc((size_t)BNt * 128 * 4);
  float* lg0   = (float*)alloc((size_t)BNt * NCls * 4);
  int*   label = (int*)alloc((size_t)BNt * 4);
  float* res   = (float*)alloc((size_t)BNt * NCls * 4);
  int*   freq  = (int*)alloc((size_t)BNt * 4);
  float* deg   = (float*)alloc((size_t)BNt * 4);
  float* dis   = (float*)alloc((size_t)BNt * 4);
  float* xw    = (float*)alloc((size_t)BNt * NCls * 4);
  float* gout  = (float*)alloc((size_t)BNt * NCls * 4);
  // bf16 staged activations (GEMM operands)
  auto allocbf = [&](size_t n) { return (__bf16*)alloc(n * 2); };
  __bf16* h64b  = allocbf((size_t)BNt * 64);
  __bf16* h128b = allocbf((size_t)BNt * 128);
  __bf16* poolb = allocbf((size_t)Bsz * 1024);
  __bf16* fcAb  = allocbf((size_t)Bsz * 512);
  __bf16* fcBb  = allocbf((size_t)Bsz * 256);
  __bf16* tkTb  = allocbf((size_t)Bsz * 16384);
  __bf16* out1b = allocbf((size_t)BNt * 64);
  __bf16* out2b = allocbf((size_t)BNt * 128);
  __bf16* out3b = allocbf((size_t)BNt * 128);
  __bf16* out4b = allocbf((size_t)BNt * 256);
  __bf16* out5b = allocbf((size_t)BNt * 512);
  __bf16* c256b = allocbf((size_t)BNt * 256);
  // bf16 weights (converted once per call)
  auto cvtW = [&](int idx, size_t n) {
    __bf16* p = allocbf(n);
    k_cvt<<<G1((long)n), 0, s>>>(F(idx), p, (long)n);
    return p;
  };
  __bf16* t3w2b  = cvtW(I_T3 + T_W2, 128 * 64);
  __bf16* t3w3b  = cvtW(I_T3 + T_W3, 1024 * 128);
  __bf16* t3fw1b = cvtW(I_T3 + T_FW1, 512 * 1024);
  __bf16* t3fw2b = cvtW(I_T3 + T_FW2, 256 * 512);
  __bf16* tkw1b  = cvtW(I_TK + T_W1, 64 * 128);
  __bf16* tkw2b  = cvtW(I_TK + T_W2, 128 * 64);
  __bf16* tkw3b  = cvtW(I_TK + T_W3, 1024 * 128);
  __bf16* tkfw1b = cvtW(I_TK + T_FW1, 512 * 1024);
  __bf16* tkfw2b = cvtW(I_TK + T_FW2, 256 * 512);
  __bf16* tkfw3b = cvtW(I_TK + T_FW3, (size_t)16384 * 256);
  __bf16* w2b    = cvtW(I_W2, 128 * 64);
  __bf16* w3b    = cvtW(I_W3, 128 * 128);
  __bf16* w4b    = cvtW(I_W4, 256 * 128);
  __bf16* w5b    = cvtW(I_W5, 512 * 256);
  __bf16* cw1b   = cvtW(I_CLS_W1, 256 * 512);
  __bf16* cw2b   = cvtW(I_CLS_W2, 128 * 256);

  const float* x   = F(I_X);
  const float* nrm = F(I_NORM);

  auto bn_relu = [&](float* act, __bf16* actb, int C, int gi, int bi, int relu, int R) {
    k_colstats<<<dim3(C), dim3(256), 0, s>>>(act, mu, var, R, C);
    k_bn_act<<<G1((long)R * C), 0, s>>>(act, actb, F(gi), F(bi), mu, var, C, (long)R * C, relu);
  };

  // ---------------- TNet(3) on x ------------------------------------------
  {
    int T = I_T3;
    k_gemm_small<true><<<G1((long)BNt * 64), 0, s>>>(x, F(T + T_W1), h64, nullptr, BNt, 64, 3, 3, 3, 64);
    bn_relu(h64, h64b, 64, T + T_BN1G, T + T_BN1B, 1, BNt);
    gemmB(s, h64b, t3w2b, h128, nullptr, nullptr, BNt, 128, 64, 64, 64, 128);
    bn_relu(h128, h128b, 128, T + T_BN2G, T + T_BN2B, 1, BNt);
    gemmB(s, h128b, t3w3b, h1024, nullptr, nullptr, BNt, 1024, 128, 128, 128, 1024);
    bn_relu(h1024, nullptr, 1024, T + T_BN3G, T + T_BN3B, 1, BNt);
    k_maxpool<<<G1(Bsz * 1024), 0, s>>>(h1024, pool, poolb, 1024);
    gemmB(s, poolb, t3fw1b, fcA, nullptr, nullptr, Bsz, 512, 1024, 1024, 1024, 512);
    bn_relu(fcA, fcAb, 512, T + T_BN4G, T + T_BN4B, 1, Bsz);
    gemmB(s, fcAb, t3fw2b, fcB, nullptr, nullptr, Bsz, 256, 512, 512, 512, 256);
    bn_relu(fcB, nullptr, 256, T + T_BN5G, T + T_BN5B, 1, Bsz);
    k_gemm_small<true><<<G1(Bsz * 9), 0, s>>>(fcB, F(T + T_FW3), t3m, F(T + T_FB3), Bsz, 9, 256, 256, 256, 9);
    k_add_eye<<<G1(Bsz * 3), 0, s>>>(t3m, 3, 9);
    k_transform3<<<G1(BNt), 0, s>>>(x, t3m, h3);
  }

  // ---------------- backbone convs 1..3 ------------------------------------
  k_gemm_small<true><<<G1((long)BNt * 64), 0, s>>>(h3, F(I_W1), out1, nullptr, BNt, 64, 3, 3, 3, 64);
  bn_relu(out1, out1b, 64, I_BN1G, I_BN1B, 1, BNt);
  gemmB(s, out1b, w2b, out2, nullptr, nullptr, BNt, 128, 64, 64, 64, 128);
  bn_relu(out2, out2b, 128, I_BN2G, I_BN2B, 1, BNt);
  gemmB(s, out2b, w3b, out3, nullptr, nullptr, BNt, 128, 128, 128, 128, 128);
  bn_relu(out3, out3b, 128, I_BN3G, I_BN3B, 1, BNt);

  // ---------------- TNet(128) on out3 --------------------------------------
  {
    int T = I_TK;
    gemmB(s, out3b, tkw1b, h64, nullptr, nullptr, BNt, 64, 128, 128, 128, 64);
    bn_relu(h64, h64b, 64, T + T_BN1G, T + T_BN1B, 1, BNt);
    gemmB(s, h64b, tkw2b, h128, nullptr, nullptr, BNt, 128, 64, 64, 64, 128);
    bn_relu(h128, h128b, 128, T + T_BN2G, T + T_BN2B, 1, BNt);
    gemmB(s, h128b, tkw3b, h1024, nullptr, nullptr, BNt, 1024, 128, 128, 128, 1024);
    bn_relu(h1024, nullptr, 1024, T + T_BN3G, T + T_BN3B, 1, BNt);
    k_maxpool<<<G1(Bsz * 1024), 0, s>>>(h1024, pool, poolb, 1024);
    gemmB(s, poolb, tkfw1b, fcA, nullptr, nullptr, Bsz, 512, 1024, 1024, 1024, 512);
    bn_relu(fcA, fcAb, 512, T + T_BN4G, T + T_BN4B, 1, Bsz);
    gemmB(s, fcAb, tkfw2b, fcB, nullptr, nullptr, Bsz, 256, 512, 512, 512, 256);
    bn_relu(fcB, fcBb, 256, T + T_BN5G, T + T_BN5B, 1, Bsz);
    gemmB(s, fcBb, tkfw3b, tkm, nullptr, F(T + T_FB3), Bsz, 16384, 256, 256, 256, 16384);
    k_add_eye<<<G1(Bsz * 128), 0, s>>>(tkm, 128, 16384);
    k_cvtT<<<G1((long)Bsz * 16384), 0, s>>>(tkm, tkTb);     // -> BT layout bf16
    // batched feature transform: h128[b] = out3[b] @ tk[b] ; fused bf16 copy
    gemmB(s, out3b, tkTb, h128, h128b, nullptr, Npt, 128, 128, 128, 128, 128,
          Bsz, (long)Npt * 128, 16384, (long)Npt * 128);
  }

  // ---------------- convs 4..5 ---------------------------------------------
  gemmB(s, h128b, w4b, out4, nullptr, nullptr, BNt, 256, 128, 128, 128, 256);
  bn_relu(out4, out4b, 256, I_BN4G, I_BN4B, 1, BNt);
  gemmB(s, out4b, w5b, out5, nullptr, nullptr, BNt, 512, 256, 256, 256, 512);
  bn_relu(out5, out5b, 512, I_BN5G, I_BN5B, 0, BNt);   // no relu on out5

  // ---------------- kNN graphs ---------------------------------------------
  k_knn3<<<G1(BNt), 0, s>>>(x, erow, ecol);     // list 1 (on xyz)
  for (int b = 0; b < Bsz; ++b) {               // list 2 (on 512-d features)
    const __bf16* Xb = out5b + (long)b * Npt * 512;
    gemmB(s, Xb, Xb, gram, nullptr, nullptr, Npt, Npt, 512, 512, 512, Npt);  // Gram
    k_sqnorm<<<G1(Npt), 0, s>>>(out5 + (long)b * Npt * 512, sq, 512);
    k_topk_gram<<<G1(Npt), 0, s>>>(gram, sq, b, erow, ecol);
  }
  k_fill_i<<<G1(E0), 0, s>>>(emask, 1, E0);
  k_mask2<<<G1(E0), 0, s>>>(erow, ecol, emask);
  k_edgecos<<<G1(2 * E0), 0, s>>>(erow, ecol, nrm, ecos, 2 * E0);

  // ---------------- classifier ---------------------------------------------
  gemmB(s, out5b, cw1b, c256, nullptr, nullptr, BNt, 256, 512, 512, 512, 256);
  bn_relu(c256, c256b, 256, I_CLS_BN1G, I_CLS_BN1B, 1, BNt);
  gemmB(s, c256b, cw2b, c128, nullptr, nullptr, BNt, 128, 256, 256, 256, 128);
  bn_relu(c128, nullptr, 128, I_CLS_BN2G, I_CLS_BN2B, 1, BNt);
  k_gemm_small<true><<<G1((long)BNt * NCls), 0, s>>>(c128, F(I_CLS_W3), lg0, F(I_CLS_B3),
                                                     BNt, NCls, 128, 128, 128, NCls);

  // ---------------- neighbor voting ----------------------------------------
  k_argmax_row<<<G1(BNt), 0, s>>>(lg0, label, NCls, BNt);
  k_fill_f<<<G1((long)BNt * NCls), 0, s>>>(res, 0.f, (long)BNt * NCls);
  k_vote<<<G1(2 * E0), 0, s>>>(erow, ecol, emask, label, res, 2 * E0);
  k_argmax_row<<<G1(BNt), 0, s>>>(res, freq, NCls, BNt);
  k_valid_w<<<G1(2 * E0), 0, s>>>(erow, ecol, emask, label, freq, ecos, ew, 2 * E0);

  // ---------------- GCN -----------------------------------------------------
  k_fill_f<<<G1(BNt), 0, s>>>(deg, 1.0f, BNt);                 // self-loop weight 1
  k_deg<<<G1(2 * E0), 0, s>>>(ecol, ew, deg, 2 * E0);
  k_dis<<<G1(BNt), 0, s>>>(deg, dis, BNt);
  k_gemm_small<false><<<G1((long)BNt * NCls), 0, s>>>(lg0, F(I_GCN_W), xw, nullptr,
                                                      BNt, NCls, NCls, NCls, NCls, NCls);
  k_gcn_init<<<G1((long)BNt * NCls), 0, s>>>(xw, dis, F(I_GCN_B), gout);
  k_gcn_edge<<<G1(2 * E0), 0, s>>>(erow, ecol, ew, dis, xw, gout, 2 * E0);

  // ---------------- outputs -------------------------------------------------
  float* O = (float*)d_out;
  const long O_OUT5 = (long)Bsz * 1088 * Npt;          // 35,651,584
  const long O_LOG  = O_OUT5 + (long)Bsz * 512 * Npt;  // 52,428,800
  const long O_PRB  = O_LOG + (long)BNt * NCls;
  const long O_LGP  = O_PRB + (long)BNt * NCls;
  // embeddings = concat([out1,out2,out3,out4,out5], axis=1) in (B,1088,N)
  k_write_bcn<<<G1((long)BNt * 64), 0, s>>>(out1, O, 64, 1088, 0);
  k_write_bcn<<<G1((long)BNt * 128), 0, s>>>(out2, O, 128, 1088, 64);
  k_write_bcn<<<G1((long)BNt * 128), 0, s>>>(out3, O, 128, 1088, 192);
  k_write_bcn<<<G1((long)BNt * 256), 0, s>>>(out4, O, 256, 1088, 320);
  k_write_bcn<<<G1((long)BNt * 512), 0, s>>>(out5, O, 512, 1088, 576);
  k_write_bcn<<<G1((long)BNt * 512), 0, s>>>(out5, O + O_OUT5, 512, 512, 0);
  k_softmax_out<<<G1(BNt), 0, s>>>(gout, O + O_LOG, O + O_PRB, O + O_LGP);
}
